// KolmogorovArnoldLinear_83288005804487
// MI455X (gfx1250) — compile-verified, matrix operand order checked by reading, not run on hardware
//
#include <hip/hip_runtime.h>

// KolmogorovArnoldLinear on MI455X (gfx1250, wave32).
// One [16384 x 5120] x [5120 x 1024] bf16 GEMM via v_wmma_f32_16x16x32_bf16
// (planes: tanh(x)^1..4 and x; pathway_balance folded into weights; the t^0
// term becomes a per-column bias). Double-buffered LDS, fragment-order
// layouts so A = 2x ds_load_b128 and B = 2x global_load_b128 per 16x16 tile.

typedef __attribute__((ext_vector_type(16))) __bf16 v16bf;
typedef __attribute__((ext_vector_type(8)))  float v8f;

union Frag {
    v16bf v;
    uint4 q[2];
};

// Hardware tanh if the builtin exists on this toolchain (gfx1250 TRANS op).
#if __has_builtin(__builtin_amdgcn_tanhf)
#define TANHF(x) __builtin_amdgcn_tanhf(x)
#elif __has_builtin(__builtin_amdgcn_tanh_f32)
#define TANHF(x) __builtin_amdgcn_tanh_f32(x)
#else
#define TANHF(x) tanhf(x)
#endif

__device__ __forceinline__ unsigned int pack_bf16(float a, float b) {
#if __has_builtin(__builtin_amdgcn_cvt_pk_bf16_f32)
    auto r = __builtin_amdgcn_cvt_pk_bf16_f32(a, b);   // v_cvt_pk_bf16_f32
    unsigned int u;
    __builtin_memcpy(&u, &r, 4);
    return u;
#else
    unsigned int ua = __builtin_bit_cast(unsigned int, a);
    unsigned int ub = __builtin_bit_cast(unsigned int, b);
    ua += 0x7fffu + ((ua >> 16) & 1u);   // RNE
    ub += 0x7fffu + ((ub >> 16) & 1u);
    return (ua >> 16) | (ub & 0xffff0000u);
#endif
}

// ---------------------------------------------------------------------------
#define IN_F   1024
#define OUT_F  1024
#define K_TOT  5120           // 5 planes * 1024
#define B_ROWS 16384          // 16 * 1024
#define TM     128
#define TN     128
#define KO_TILES (K_TOT / 32) // 160
#define NO_TILES (OUT_F / 16) // 64
#define WPACK_BYTES ((size_t)KO_TILES * NO_TILES * 1024)  // 10,485,760
#define PLANE_B  8192u        // one bf16 [128 x 32] plane in fragment order
#define BUF_B    (5u * PLANE_B)
#define KT_ROW_B 65536u       // bytes per kt row: 64 tiles * 1024 B
#define P_STRIDE (32u * KT_ROW_B)   // 2 MB between planes in wpack

// ---------------------------------------------------------------------------
// Pack weights into per-lane WMMA B-fragment order (bf16).
// Wsrc[k][o]:
//   k in [0,4096):    pb0 * basis_coeffs[i=k%1024][o][n=k/1024 + 1]
//   k in [4096,5120): pb1 * linear_w[o][i=k-4096]
// B-fragment (32x16 tile): lane L holds N=L&15, half h=L>>4; dword j holds
// K pair: j<4 -> K0=2j+8h ; j>=4 -> K0=16+2(j-4)+8h. 32 B/lane, contiguous.
// ---------------------------------------------------------------------------
__global__ void kal_pack_w(const float* __restrict__ bc,
                           const float* __restrict__ lw,
                           const float* __restrict__ pb,
                           unsigned int* __restrict__ wpack) {
    const int tile = blockIdx.x;        // 0 .. 160*64-1
    const int ko   = tile >> 6;
    const int no   = tile & 63;
    const int lane = threadIdx.x;
    const int h    = lane >> 4;
    const int o    = no * 16 + (lane & 15);
    const float pb0 = pb[0], pb1 = pb[1];

    unsigned int u[8];
#pragma unroll
    for (int j = 0; j < 8; ++j) {
        const int K0 = (j < 4) ? (2 * j + 8 * h) : (16 + 2 * (j - 4) + 8 * h);
        float v01[2];
#pragma unroll
        for (int e = 0; e < 2; ++e) {
            const int kglob = ko * 32 + K0 + e;
            float v;
            if (kglob < 4 * IN_F) {
                const int nb = (kglob >> 10) + 1;       // 1..4
                const int i  = kglob & (IN_F - 1);
                v = pb0 * bc[(size_t)i * (OUT_F * 5) + o * 5 + nb];
            } else {
                const int i = kglob - 4 * IN_F;
                v = pb1 * lw[(size_t)o * IN_F + i];
            }
            v01[e] = v;
        }
        u[j] = pack_bf16(v01[0], v01[1]);
    }
    unsigned int* dst = wpack + (size_t)tile * 256 + lane * 8;
#pragma unroll
    for (int j = 0; j < 8; ++j) dst[j] = u[j];
}

// bias[o] = pb0 * sum_i basis_coeffs[i][o][0]   (the t^0 term)
__global__ void kal_bias(const float* __restrict__ bc,
                         const float* __restrict__ pb,
                         float* __restrict__ bias) {
    const int o = blockIdx.x * blockDim.x + threadIdx.x;
    if (o < OUT_F) {
        float s = 0.f;
        for (int i = 0; i < IN_F; ++i)
            s += bc[(size_t)i * (OUT_F * 5) + o * 5];
        bias[o] = pb[0] * s;
    }
}

// ---------------------------------------------------------------------------
// Fused tanh/power + GEMM. 256 threads = 8 wave32 -> 128x128 C tile.
// Waves 2(M) x 4(N); each wave 4x2 sub-tiles of 16x16 -> 40 WMMA / K-step.
// ---------------------------------------------------------------------------
__global__ __launch_bounds__(256) void kal_gemm(const float* __restrict__ x,
                                                const unsigned char* __restrict__ wpb,
                                                const float* __restrict__ bias,
                                                float* __restrict__ out) {
    __shared__ __align__(16) unsigned char lds[2 * BUF_B];   // 80 KB, double buf

    const int tid    = threadIdx.x;
    const int lane   = tid & 31;
    const int w      = tid >> 5;
    const int wave_m = w >> 2;            // 0..1
    const int wave_n = w & 3;             // 0..3
    const int nTile  = blockIdx.x;        // fast dim: 8 blocks share x rows
    const int mBase  = blockIdx.y * TM;

    // Producer mapping: 2 threads/row, 16 consecutive K each.
    // Even thread (k=0..15): h0 j0..3 of lane rr, then h1 j0..3 of lane rr+16.
    // Odd  thread (k=16..31): h0 j4..7 of lane rr, then h1 j4..7 of rr+16.
    // Either way: b128 #0 = f[0..7] at off, b128 #1 = f[8..15] at off+512.
    const int pr  = tid >> 1;
    const unsigned prodOff = (unsigned)((pr >> 4) * 1024 + (pr & 15) * 32 +
                                        (tid & 1) * 16);
    const float* xptr = x + (size_t)(mBase + pr) * IN_F + (tid & 1) * 16;

    // Running B-fragment pointers (per-lane, 32 B/fragment), +64 KB per K-step.
    const unsigned char* gpA = wpb +
        ((size_t)(nTile * 8 + wave_n * 2) * 1024 + lane * 32);   // planes 0..3
    const unsigned char* gpB = gpA + (size_t)4 * P_STRIDE;       // plane 4

    v8f acc[4][2];
#pragma unroll
    for (int ms = 0; ms < 4; ++ms)
#pragma unroll
        for (int ns = 0; ns < 2; ++ns)
#pragma unroll
            for (int r = 0; r < 8; ++r) acc[ms][ns][r] = 0.f;

    auto produce = [&](int buf) {
        const float4* s4 = reinterpret_cast<const float4*>(xptr);
        float4 a0 = s4[0], a1 = s4[1], a2 = s4[2], a3 = s4[3];
        xptr += 32;
        float f[16] = {a0.x, a0.y, a0.z, a0.w, a1.x, a1.y, a1.z, a1.w,
                       a2.x, a2.y, a2.z, a2.w, a3.x, a3.y, a3.z, a3.w};
        unsigned char* dst = lds + buf * BUF_B + prodOff;
        // plane 4: raw x (linear pathway)
        {
            uint4 u0, u1;
            u0.x = pack_bf16(f[0], f[1]);   u0.y = pack_bf16(f[2], f[3]);
            u0.z = pack_bf16(f[4], f[5]);   u0.w = pack_bf16(f[6], f[7]);
            u1.x = pack_bf16(f[8], f[9]);   u1.y = pack_bf16(f[10], f[11]);
            u1.z = pack_bf16(f[12], f[13]); u1.w = pack_bf16(f[14], f[15]);
            *(uint4*)(dst + 4 * PLANE_B)       = u0;
            *(uint4*)(dst + 4 * PLANE_B + 512) = u1;
        }
        float t[16], c[16];
#pragma unroll
        for (int e = 0; e < 16; ++e) { t[e] = TANHF(f[e]); c[e] = t[e]; }
#pragma unroll
        for (int p = 0; p < 4; ++p) {       // t, t^2, t^3, t^4
            uint4 u0, u1;
            u0.x = pack_bf16(c[0], c[1]);   u0.y = pack_bf16(c[2], c[3]);
            u0.z = pack_bf16(c[4], c[5]);   u0.w = pack_bf16(c[6], c[7]);
            u1.x = pack_bf16(c[8], c[9]);   u1.y = pack_bf16(c[10], c[11]);
            u1.z = pack_bf16(c[12], c[13]); u1.w = pack_bf16(c[14], c[15]);
            *(uint4*)(dst + p * PLANE_B)       = u0;
            *(uint4*)(dst + p * PLANE_B + 512) = u1;
            if (p < 3) {
#pragma unroll
                for (int e = 0; e < 16; ++e) c[e] *= t[e];
            }
        }
    };

    auto consume = [&](int buf, const unsigned char* gA, const unsigned char* gB) {
        const unsigned char* sb =
            lds + buf * BUF_B + (wave_m * 4) * 1024 + lane * 32;
#pragma unroll
        for (int p = 0; p < 5; ++p) {
            Frag a[4], b[2];
#pragma unroll
            for (int ms = 0; ms < 4; ++ms) {
                const uint4* sp = (const uint4*)(sb + p * PLANE_B + ms * 1024);
                a[ms].q[0] = sp[0];
                a[ms].q[1] = sp[1];
            }
            const unsigned char* g = (p < 4) ? (gA + p * P_STRIDE) : gB;
#pragma unroll
            for (int ns = 0; ns < 2; ++ns) {
                const uint4* gp = (const uint4*)(g + ns * 1024);
                b[ns].q[0] = gp[0];
                b[ns].q[1] = gp[1];
            }
#pragma unroll
            for (int ms = 0; ms < 4; ++ms)
#pragma unroll
                for (int ns = 0; ns < 2; ++ns)
                    acc[ms][ns] = __builtin_amdgcn_wmma_f32_16x16x32_bf16(
                        false, a[ms].v, false, b[ns].v,
                        (short)0, acc[ms][ns], false, false);
        }
    };

    produce(0);
#pragma unroll 1
    for (int it = 0; it < 32; ++it) {
        __syncthreads();
        if (it + 1 < 32) produce((it + 1) & 1);   // overlap loads+tanh w/ WMMA
        consume(it & 1, gpA, gpB);
        gpA += KT_ROW_B;
        gpB += KT_ROW_B;
    }

    // Epilogue: C layout -> VGPR r holds M = r + 8*(lane>>4), N = lane&15
    const int h  = lane >> 4;
    const int cl = lane & 15;
#pragma unroll
    for (int ns = 0; ns < 2; ++ns) {
        const int col = nTile * TN + wave_n * 32 + ns * 16 + cl;
        const float bv = bias[col];
#pragma unroll
        for (int ms = 0; ms < 4; ++ms) {
            const int rowBase = mBase + wave_m * 64 + ms * 16 + 8 * h;
#pragma unroll
            for (int r = 0; r < 8; ++r)
                out[(size_t)(rowBase + r) * OUT_F + col] = acc[ms][ns][r] + bv;
        }
    }
}

// ---------------------------------------------------------------------------
extern "C" void kernel_launch(void* const* d_in, const int* in_sizes, int n_in,
                              void* d_out, int out_size, void* d_ws, size_t ws_size,
                              hipStream_t stream) {
    (void)in_sizes; (void)n_in; (void)out_size; (void)ws_size;
    const float* x  = (const float*)d_in[0];  // [16,1024,1024]
    const float* bc = (const float*)d_in[1];  // [1024,1024,5]
    const float* lw = (const float*)d_in[2];  // [1024,1024]
    const float* pb = (const float*)d_in[3];  // [2]
    float* out = (float*)d_out;               // [16,1024,1024]

    unsigned int* wpack = (unsigned int*)d_ws;                 // 10 MB bf16 frags
    float* bias = (float*)((char*)d_ws + WPACK_BYTES);         // 4 KB

    kal_pack_w<<<KO_TILES * NO_TILES, 32, 0, stream>>>(bc, lw, pb, wpack);
    kal_bias<<<OUT_F / 256, 256, 0, stream>>>(bc, pb, bias);

    dim3 grid(OUT_F / TN, B_ROWS / TM);   // (8 N-tiles fast, 128 M-tiles)
    kal_gemm<<<grid, 256, 0, stream>>>(
        x, (const unsigned char*)wpack, bias, out);
}
